// OT_GNN_layer_8675833938655
// MI455X (gfx1250) — compile-verified, matrix operand order checked by reading, not run on hardware
//
#include <hip/hip_runtime.h>
#include <hip/hip_bf16.h>

// ---- problem constants (match reference setup) ----
#define FDIM   128
#define TCNT   16
#define TN     8
#define KNEIGH 16
#define NLOC   17          // star subgraph size = 1 + KNEIGH
#define N_OUTER 5
#define N_INNER 10
#define INV_EPS 5.0f       // 1/0.2
#define TINY_V  1e-30f

typedef __attribute__((ext_vector_type(16))) __bf16 v16bf;
typedef __attribute__((ext_vector_type(8)))  float  v8f;

// ---------------- wave32 butterfly reductions ----------------
__device__ __forceinline__ float wsum32(float x) {
#pragma unroll
    for (int m = 16; m >= 1; m >>= 1) x += __shfl_xor(x, m, 32);
    return x;
}
__device__ __forceinline__ float wmin32(float x) {
#pragma unroll
    for (int m = 16; m >= 1; m >>= 1) x = fminf(x, __shfl_xor(x, m, 32));
    return x;
}

// ---------------- kernel 0: squared row norms ----------------
// one wave per row of a [rows x 128] fp32 matrix
__global__ __launch_bounds__(256) void rownorm_kernel(
    const float* __restrict__ src, float* __restrict__ dst, int rows) {
    const int row  = blockIdx.x * 8 + (threadIdx.x >> 5);
    const int lane = threadIdx.x & 31;
    if (row >= rows) return;
    const float* p = src + (size_t)row * FDIM;
    float s = 0.f;
#pragma unroll
    for (int j = 0; j < FDIM / 32; ++j) {
        float v = p[lane + j * 32];
        s = fmaf(v, v, s);
    }
    s = wsum32(s);
    if (lane == 0) dst[row] = s;
}

// ---------------- kernel 1: G = scaled distance GEMM (WMMA) ----------------
// G[r,c] = (xn2[r] + tn2[c] - 2 * x_r . tf_c) / 128,  c = t*8+m
// block = 256 threads = 8 waves; each wave owns a 16-row x 128-col tile.
// Split-bf16: x ~ hi+lo; acc += Ahi*Bhi + Ahi*Blo + Alo*Bhi (f32 accum).
__global__ __launch_bounds__(256) void gemm_dist_kernel(
    const float* __restrict__ x, const float* __restrict__ tf,
    const float* __restrict__ xn2, const float* __restrict__ tn2,
    float* __restrict__ G, int N) {
    const int lane    = threadIdx.x & 31;
    const int wave    = threadIdx.x >> 5;
    const int rowBase = (blockIdx.x * 8 + wave) * 16;
    const int mr      = lane & 15;   // M (A) / N (B) index within 16x16 tile
    const int kh      = lane >> 4;   // K-half selector per ISA 16-bit layout

    int arow = rowBase + mr;
    if (arow >= N) arow = N - 1;                 // clamp; stores are guarded
    const float* xrow = x + (size_t)arow * FDIM;

    v8f acc[8] = {};                              // 8 N-tiles of 16x16 f32

#pragma unroll
    for (int kc = 0; kc < 4; ++kc) {              // K chunks of 32
        const int kbase = kc * 32 + kh * 8;
        v16bf Ahi, Alo;
#pragma unroll
        for (int j = 0; j < 8; ++j) {
            float a0 = xrow[kbase + j];
            float a1 = xrow[kbase + 16 + j];
            __bf16 h0 = (__bf16)a0, h1 = (__bf16)a1;
            Ahi[j]     = h0;  Alo[j]     = (__bf16)(a0 - (float)h0);
            Ahi[8 + j] = h1;  Alo[8 + j] = (__bf16)(a1 - (float)h1);
        }
#pragma unroll
        for (int nt = 0; nt < 8; ++nt) {
            const float* trow = tf + (size_t)(nt * 16 + mr) * FDIM;
            v16bf Bhi, Blo;
#pragma unroll
            for (int j = 0; j < 8; ++j) {
                float b0 = trow[kbase + j];
                float b1 = trow[kbase + 16 + j];
                __bf16 h0 = (__bf16)b0, h1 = (__bf16)b1;
                Bhi[j]     = h0;  Blo[j]     = (__bf16)(b0 - (float)h0);
                Bhi[8 + j] = h1;  Blo[8 + j] = (__bf16)(b1 - (float)h1);
            }
            acc[nt] = __builtin_amdgcn_wmma_f32_16x16x32_bf16(
                false, Ahi, false, Bhi, (short)0, acc[nt], false, false);
            acc[nt] = __builtin_amdgcn_wmma_f32_16x16x32_bf16(
                false, Ahi, false, Blo, (short)0, acc[nt], false, false);
            acc[nt] = __builtin_amdgcn_wmma_f32_16x16x32_bf16(
                false, Alo, false, Bhi, (short)0, acc[nt], false, false);
        }
    }

    // D layout: VGPR r, lanes 0-15 -> M=r, lanes 16-31 -> M=r+8; N = lane&15
#pragma unroll
    for (int nt = 0; nt < 8; ++nt) {
        const int col = nt * 16 + mr;
        const float t2 = tn2[col];
#pragma unroll
        for (int r = 0; r < 8; ++r) {
            const int grow = rowBase + r + kh * 8;
            if (grow < N)
                G[(size_t)grow * FDIM + col] =
                    (xn2[grow] + t2 - 2.0f * acc[nt][r]) * (1.0f / (float)FDIM);
        }
    }
}

// ---------------- kernel 2: per-(node,template) entropic FGW solve ----------------
// block = 512 threads = 16 waves; blockIdx.x = node; wave = template.
// lane = plan row a (0..16 active); 8 plan columns held in registers.
__global__ __launch_bounds__(512) void fgw_kernel(
    const int* __restrict__ dst_idx,     // edge_index dst, [N*16]
    const float* __restrict__ lt,        // latent_template [16,8,8]
    const float* __restrict__ G,         // [N,128]
    const float* __restrict__ W,         // [16,8]
    const float* __restrict__ bvec,      // [8]
    float* __restrict__ out, int N) {
    __shared__ float C2sh[TCNT * 64];
    __shared__ float fgw_sh[TCNT];

    const int i    = blockIdx.x;
    const int t    = threadIdx.x >> 5;
    const int lane = threadIdx.x & 31;
    float* C2t = C2sh + t * 64;

    // row index in x/G for this lane (center or neighbor); prefetch its G slice
    int r = i;
    if (lane >= 1 && lane < NLOC) r = dst_idx[i * KNEIGH + (lane - 1)];
    const float* grow = G + (size_t)r * FDIM + t * TN;
    __builtin_prefetch(grow, 0, 1);

    // stage symmetrized template structure C2^t = 0.5*(L + L^T) into LDS
#pragma unroll
    for (int rep = 0; rep < 2; ++rep) {
        int idx = lane + rep * 32;
        int l = idx >> 3, b2 = idx & 7;
        C2t[idx] = 0.5f * (lt[t * 64 + l * 8 + b2] + lt[t * 64 + b2 * 8 + l]);
    }
    __syncthreads();

    const bool  act  = (lane < NLOC);
    const float hrow = act ? (1.0f / (float)NLOC) : 0.0f;
    const float c1h  = (lane == 0) ? ((float)KNEIGH / (float)NLOC)
                                   : (act ? (1.0f / (float)NLOC) : 0.0f);

    float Mr[TN], cC[TN], P[TN], Km[TN], vv[TN];
#pragma unroll
    for (int m = 0; m < TN; ++m) {
        // c2p[t][m] = (1/8) * sum_j C2[m,j]^2
        float q = 0.f;
#pragma unroll
        for (int j = 0; j < TN; ++j) { float c = C2t[m * 8 + j]; q = fmaf(c, c, q); }
        q *= 0.125f;
        Mr[m] = act ? grow[m] : 0.0f;
        cC[m] = act ? (c1h + q) : 0.0f;
        P[m]  = act ? (1.0f / (float)(NLOC * TN)) : 0.0f;   // h[a]*p[m]
    }

    for (int outer = 0; outer < N_OUTER; ++outer) {
        // S = C1 @ P : row0 = sum of rows 1..16; rows>=1 = row0 of P
        float S[TN];
#pragma unroll
        for (int m = 0; m < TN; ++m) {
            float tot = wsum32(P[m]);
            float p0  = __shfl(P[m], 0, 32);
            S[m] = (lane == 0) ? (tot - p0) : p0;
        }
        // grad = (1-a)M + a(constC - 2 * S@C2); stabilize by tile min
        float gmin = 1e30f;
#pragma unroll
        for (int m = 0; m < TN; ++m) {
            float A = 0.f;
#pragma unroll
            for (int j = 0; j < TN; ++j) A = fmaf(S[j], C2t[j * 8 + m], A);
            float g = 0.5f * Mr[m] + 0.5f * (cC[m] - 2.0f * A);
            Km[m] = g;
            if (act) gmin = fminf(gmin, g);
        }
        gmin = wmin32(gmin);
#pragma unroll
        for (int m = 0; m < TN; ++m)
            Km[m] = P[m] * __expf((gmin - Km[m]) * INV_EPS);

        // Sinkhorn
#pragma unroll
        for (int m = 0; m < TN; ++m) vv[m] = 1.0f;
        float u = 0.0f;
        for (int it = 0; it < N_INNER; ++it) {
            float s = 0.f;
#pragma unroll
            for (int m = 0; m < TN; ++m) s = fmaf(Km[m], vv[m], s);
            u = hrow / (s + TINY_V);
#pragma unroll
            for (int m = 0; m < TN; ++m) {
                float c = wsum32(Km[m] * u);
                vv[m] = (1.0f / (float)TN) / (c + TINY_V);
            }
        }
#pragma unroll
        for (int m = 0; m < TN; ++m) P[m] = u * Km[m] * vv[m];
    }

    // final FGW value with the converged plan
    float S[TN];
#pragma unroll
    for (int m = 0; m < TN; ++m) {
        float tot = wsum32(P[m]);
        float p0  = __shfl(P[m], 0, 32);
        S[m] = (lane == 0) ? (tot - p0) : p0;
    }
    float loc = 0.f;
#pragma unroll
    for (int m = 0; m < TN; ++m) {
        float A = 0.f;
#pragma unroll
        for (int j = 0; j < TN; ++j) A = fmaf(S[j], C2t[j * 8 + m], A);
        loc = fmaf(0.5f * Mr[m] + 0.5f * (cC[m] - 2.0f * A), P[m], loc);
    }
    float f = wsum32(loc);
    if (lane == 0) fgw_sh[t] = f;
    __syncthreads();

    // out[i,:] = fgw @ W + b  (wave 0, lanes 0..7)
    if (t == 0 && lane < 8) {
        float o = bvec[lane];
#pragma unroll
        for (int tt = 0; tt < TCNT; ++tt) o = fmaf(fgw_sh[tt], W[tt * 8 + lane], o);
        out[(size_t)i * 8 + lane] = o;
    }
}

// ---------------- host-side launcher ----------------
extern "C" void kernel_launch(void* const* d_in, const int* in_sizes, int n_in,
                              void* d_out, int out_size, void* d_ws, size_t ws_size,
                              hipStream_t stream) {
    const float* x   = (const float*)d_in[0];   // [N,128]
    const int*   ei  = (const int*)  d_in[1];   // [2, N*16]
    const float* lt  = (const float*)d_in[2];   // [16,8,8]
    const float* tf  = (const float*)d_in[3];   // [16,8,128] == [128,128]
    const float* W   = (const float*)d_in[4];   // [16,8]
    const float* bv  = (const float*)d_in[5];   // [8]
    float* out = (float*)d_out;

    const int N = in_sizes[0] / FDIM;           // 10000
    const int* dst_idx = ei + (size_t)N * KNEIGH;

    // workspace: G [N*128] f32, xn2 [N], tn2 [128]  (~5.2 MB)
    float* G   = (float*)d_ws;
    float* xn2 = G + (size_t)N * FDIM;
    float* tn2 = xn2 + N;

    rownorm_kernel<<<(N + 7) / 8, 256, 0, stream>>>(x, xn2, N);
    rownorm_kernel<<<(128 + 7) / 8, 256, 0, stream>>>(tf, tn2, 128);
    gemm_dist_kernel<<<(N + 127) / 128, 256, 0, stream>>>(x, tf, xn2, tn2, G, N);
    fgw_kernel<<<N, 512, 0, stream>>>(dst_idx, lt, G, W, bv, out, N);
}
